// CausalSelfAttention_58454504898937
// MI455X (gfx1250) — matrix operand order repaired and checked by reading, not compiled, not run on hardware
//
#include <hip/hip_runtime.h>
#include <hip/hip_bf16.h>

typedef _Float16 v8h  __attribute__((ext_vector_type(8)));
typedef _Float16 v16h __attribute__((ext_vector_type(16)));
typedef float    v8f  __attribute__((ext_vector_type(8)));

#define T_SEQ 2048
#define C_EMB 4096
#define HS 128
#define NH 32
#define NG 8
#define QKV_DIM 6144      // (32 + 2*8) * 128
#define GROUP_STRIDE 768  // 6 * 128

__device__ __forceinline__ v8f wmma16(v16h a, v16h b, v8f c) {
  // D(16x16 f32) = A(16x32 f16) * B(32x16 f16) + C
  return __builtin_amdgcn_wmma_f32_16x16x32_f16(false, a, false, b, (short)0, c,
                                                false, false);
}

// concat two v8h (elements 0..7 = lo, 8..15 = hi)
__device__ __forceinline__ v16h cat8(v8h lo, v8h hi) {
  return __builtin_shufflevector(lo, hi, 0, 1, 2, 3, 4, 5, 6, 7, 8, 9, 10, 11,
                                 12, 13, 14, 15);
}

__device__ __forceinline__ float rmax16(float v) {
  v = fmaxf(v, __shfl_xor(v, 1, 32));
  v = fmaxf(v, __shfl_xor(v, 2, 32));
  v = fmaxf(v, __shfl_xor(v, 4, 32));
  v = fmaxf(v, __shfl_xor(v, 8, 32));
  return v;
}
__device__ __forceinline__ float rsum16(float v) {
  v += __shfl_xor(v, 1, 32);
  v += __shfl_xor(v, 2, 32);
  v += __shfl_xor(v, 4, 32);
  v += __shfl_xor(v, 8, 32);
  return v;
}

// ---------------------------------------------------------------------------
// f32 -> f16 conversion, grid-stride
// ---------------------------------------------------------------------------
__global__ void cvt_f32_to_f16(const float* __restrict__ in,
                               _Float16* __restrict__ out, int n) {
  int i = blockIdx.x * blockDim.x + threadIdx.x;
  const int stride = gridDim.x * blockDim.x;
  for (; i < n; i += stride) out[i] = (_Float16)in[i];
}

// ---------------------------------------------------------------------------
// C[M,N] = A[M,K] * B[N,K]^T  (A,B fp16, K contiguous; C fp32)
// Block: 128 threads (4 waves 2x2), block tile 128x128, wave tile 64x64.
// A frag (16x32): lane m=lid needs K = k0+half*16+0..15 (contiguous 32B).
// B frag (32x16): lane n=lid needs K = k0+half*16+0..15 (contiguous 32B).
// ---------------------------------------------------------------------------
__global__ __launch_bounds__(128) void gemm_nt_f16(
    const _Float16* __restrict__ A, const _Float16* __restrict__ B,
    float* __restrict__ C, int M, int N, int K) {
  const int lane = threadIdx.x & 31;
  const int wave = threadIdx.x >> 5;
  const int lid  = lane & 15;
  const int half = lane >> 4;
  const int row0 = blockIdx.y * 128 + (wave >> 1) * 64;
  const int col0 = blockIdx.x * 128 + (wave & 1) * 64;

  v8f acc[4][4] = {};
  const _Float16* Ap = A + (size_t)(row0 + lid) * K;
  const _Float16* Bp = B + (size_t)(col0 + lid) * K + half * 16;

  for (int k0 = 0; k0 < K; k0 += 32) {
    v16h a[4];
#pragma unroll
    for (int i = 0; i < 4; ++i) {
      const _Float16* ar = Ap + (size_t)i * 16 * K + k0 + half * 16;
      a[i] = cat8(*(const v8h*)ar, *(const v8h*)(ar + 8));
    }
#pragma unroll
    for (int j = 0; j < 4; ++j) {
      v16h b = *(const v16h*)(Bp + (size_t)j * 16 * K + k0);
#pragma unroll
      for (int i = 0; i < 4; ++i) acc[i][j] = wmma16(a[i], b, acc[i][j]);
    }
  }

#pragma unroll
  for (int i = 0; i < 4; ++i)
#pragma unroll
    for (int j = 0; j < 4; ++j)
#pragma unroll
      for (int r = 0; r < 8; ++r)
        C[(size_t)(row0 + i * 16 + r + half * 8) * N +
          (col0 + j * 16 + lid)] = acc[i][j][r];
}

// ---------------------------------------------------------------------------
// Fused RoPE + f32->f16 split of the qkv buffer.
//   slots 0..3 (q): rope, fold 1/sqrt(128), -> qh[h][t][d]
//   slot 4 (k):     rope                    -> kh[g][t][d]
//   slot 5 (v):     copy TRANSPOSED        -> vth[g][d][t]   (feeds PV B-frags)
// grid (T, NG*6), block 64 (lane d handles the rope pair (d, d+64)).
// ---------------------------------------------------------------------------
__global__ void rope_split_kernel(const float* __restrict__ qkv,
                                  const float* __restrict__ cosb,
                                  const float* __restrict__ sinb,
                                  _Float16* __restrict__ qh,
                                  _Float16* __restrict__ kh,
                                  _Float16* __restrict__ vth) {
  const int d = threadIdx.x;  // 0..63
  const int t = blockIdx.x;
  const int s = blockIdx.y;   // 0..47
  const int g = s / 6, slot = s % 6;
  const size_t base = (size_t)t * QKV_DIM + g * GROUP_STRIDE + slot * HS;
  const float x1 = qkv[base + d];
  const float x2 = qkv[base + d + 64];

  if (slot == 5) {  // V, transposed store
    vth[((size_t)g * HS + d) * T_SEQ + t]      = (_Float16)x1;
    vth[((size_t)g * HS + d + 64) * T_SEQ + t] = (_Float16)x2;
    return;
  }
  const float c  = cosb[t * HS + d];
  const float sn = sinb[t * HS + d];
  const float r1 = x1 * c - x2 * sn;
  const float r2 = x2 * c + x1 * sn;
  if (slot == 4) {  // K
    kh[((size_t)g * T_SEQ + t) * HS + d]      = (_Float16)r1;
    kh[((size_t)g * T_SEQ + t) * HS + d + 64] = (_Float16)r2;
  } else {          // Q, scale folded
    const int h = g * 4 + slot;
    const float scale = 0.08838834764831845f;  // 1/sqrt(128)
    qh[((size_t)h * T_SEQ + t) * HS + d]      = (_Float16)(r1 * scale);
    qh[((size_t)h * T_SEQ + t) * HS + d + 64] = (_Float16)(r2 * scale);
  }
}

// ---------------------------------------------------------------------------
// Causal flash attention, fp16 WMMA / fp32 softmax+accum.
// One wave per (head, 16-query tile); 64-key tiles per iteration so that the
// per-tile VALU overhead (shuffle reductions, alpha rescale, LDS round-trip)
// is amortized over twice as many keys. 32 WMMA per tile (16 QK^T + 16 PV).
// grid (T/64, NH), block 128.
// ---------------------------------------------------------------------------
__global__ __launch_bounds__(128) void flash_f16(
    const _Float16* __restrict__ qh, const _Float16* __restrict__ kh,
    const _Float16* __restrict__ vth, _Float16* __restrict__ yh) {
  __shared__ __align__(16) _Float16 smem[4 * 16 * 72];  // per-wave 16x(64+8)
  const int lane = threadIdx.x & 31;
  const int wave = threadIdx.x >> 5;
  const int lid  = lane & 15;
  const int half = lane >> 4;
  const int h  = blockIdx.y;
  const int qt = blockIdx.x * 4 + wave;  // 0..127
  const int g  = h >> 2;
  const int q0 = qt * 16;
  _Float16* lds = smem + wave * (16 * 72);

  // Preload Q A-fragments (scale already folded in)
  v16h qf[4];
  {
    const _Float16* qrow = qh + ((size_t)h * T_SEQ + q0 + lid) * HS;
#pragma unroll
    for (int c = 0; c < 4; ++c) {
      const _Float16* p = qrow + c * 32 + half * 16;
      qf[c] = cat8(*(const v8h*)p, *(const v8h*)(p + 8));
    }
  }

  v8f o[8] = {};
  float mi[8], li[8];
#pragma unroll
  for (int r = 0; r < 8; ++r) { mi[r] = -3.0e38f; li[r] = 0.0f; }

  const int ntiles = qt / 4 + 1;  // 64-key tiles covering keys <= q0+15
  for (int j = 0; j < ntiles; ++j) {
    // S(16x64) = Q K^T  as four 16x16 C-tiles
    v8f s[4] = {};
    const _Float16* kbase =
        kh + ((size_t)g * T_SEQ + j * 64 + lid) * HS + half * 16;
#pragma unroll
    for (int c = 0; c < 4; ++c) {
#pragma unroll
      for (int u = 0; u < 4; ++u) {
        const v16h b = *(const v16h*)(kbase + (size_t)u * 16 * HS + c * 32);
        s[u] = wmma16(qf[c], b, s[u]);
      }
    }

    // Online softmax; rows r / r+8 live in lane halves 0/1 of component r
    float alpha[8];
#pragma unroll
    for (int r = 0; r < 8; ++r) {
      const int row = q0 + r + half * 8;
      float sv[4];
      float mx = -3.0e38f;
#pragma unroll
      for (int u = 0; u < 4; ++u) {
        const int col = j * 64 + u * 16 + lid;
        sv[u] = (row >= col) ? s[u][r] : -3.0e38f;
        mx = fmaxf(mx, sv[u]);
      }
      const float nm = fmaxf(mi[r], rmax16(mx));
      alpha[r] = __expf(mi[r] - nm);
      mi[r] = nm;
      float psum = 0.0f;
#pragma unroll
      for (int u = 0; u < 4; ++u) {
        const float p = __expf(sv[u] - nm);
        psum += p;
        lds[(r + half * 8) * 72 + u * 16 + lid] = (_Float16)p;
      }
      li[r] = li[r] * alpha[r] + rsum16(psum);
    }
#pragma unroll
    for (int n = 0; n < 8; ++n)
#pragma unroll
      for (int r = 0; r < 8; ++r) o[n][r] *= alpha[r];

    __asm__ volatile("s_wait_dscnt 0" ::: "memory");

    // O += P * V over two 32-key chunks; P A-frags from LDS transpose,
    // V^T rows give contiguous key-major B-fragments.
    const _Float16* pr = lds + lid * 72;
    const _Float16* vp =
        vth + ((size_t)g * HS + lid) * T_SEQ + j * 64 + half * 16;
#pragma unroll
    for (int cc = 0; cc < 2; ++cc) {
      const _Float16* pc = pr + cc * 32 + half * 16;
      const v16h pa = cat8(*(const v8h*)pc, *(const v8h*)(pc + 8));
#pragma unroll
      for (int n = 0; n < 8; ++n) {
        const v16h vb = *(const v16h*)(vp + (size_t)n * 16 * T_SEQ + cc * 32);
        o[n] = wmma16(pa, vb, o[n]);
      }
    }
    __asm__ volatile("s_wait_dscnt 0" ::: "memory");
  }

  // Normalize, store y[t, h*128 + d] as fp16 (input to proj GEMM)
#pragma unroll
  for (int n = 0; n < 8; ++n)
#pragma unroll
    for (int r = 0; r < 8; ++r)
      yh[(size_t)(q0 + r + half * 8) * (NH * HS) + h * HS + n * 16 + lid] =
          (_Float16)(o[n][r] / li[r]);
}

// ---------------------------------------------------------------------------
extern "C" void kernel_launch(void* const* d_in, const int* in_sizes, int n_in,
                              void* d_out, int out_size, void* d_ws,
                              size_t ws_size, hipStream_t stream) {
  const float* x      = (const float*)d_in[0];
  const float* cosb   = (const float*)d_in[1];
  const float* sinb   = (const float*)d_in[2];
  const float* w_attn = (const float*)d_in[3];
  const float* w_proj = (const float*)d_in[4];
  float* out = (float*)d_out;

  // Workspace carve-up (~193 MB total)
  char* w = (char*)d_ws;
  size_t off = 0;
  float* qkv = (float*)(w + off);       off += (size_t)T_SEQ * QKV_DIM * 4;
  _Float16* xh  = (_Float16*)(w + off); off += (size_t)T_SEQ * C_EMB * 2;
  _Float16* wah = (_Float16*)(w + off); off += (size_t)QKV_DIM * C_EMB * 2;
  _Float16* wph = (_Float16*)(w + off); off += (size_t)C_EMB * C_EMB * 2;
  _Float16* qhp = (_Float16*)(w + off); off += (size_t)NH * T_SEQ * HS * 2;
  _Float16* khp = (_Float16*)(w + off); off += (size_t)NG * T_SEQ * HS * 2;
  _Float16* vtp = (_Float16*)(w + off); off += (size_t)NG * T_SEQ * HS * 2;
  _Float16* yh  = (_Float16*)(w + off); off += (size_t)T_SEQ * C_EMB * 2;

  // 0) convert inputs to fp16
  cvt_f32_to_f16<<<2048, 256, 0, stream>>>(x, xh, T_SEQ * C_EMB);
  cvt_f32_to_f16<<<4096, 256, 0, stream>>>(w_attn, wah, QKV_DIM * C_EMB);
  cvt_f32_to_f16<<<4096, 256, 0, stream>>>(w_proj, wph, C_EMB * C_EMB);

  // 1) qkv = x @ w_attn^T (fp16 in, fp32 out)
  gemm_nt_f16<<<dim3(QKV_DIM / 128, T_SEQ / 128), dim3(128), 0, stream>>>(
      xh, wah, qkv, T_SEQ, QKV_DIM, C_EMB);

  // 2) RoPE + split to fp16 q/k/v^T
  rope_split_kernel<<<dim3(T_SEQ, NG * 6), dim3(64), 0, stream>>>(
      qkv, cosb, sinb, qhp, khp, vtp);

  // 3) causal flash attention -> y (fp16)
  flash_f16<<<dim3(T_SEQ / 64, NH), dim3(128), 0, stream>>>(qhp, khp, vtp, yh);

  // 4) out = y @ w_proj^T (fp32 result)
  gemm_nt_f16<<<dim3(C_EMB / 128, T_SEQ / 128), dim3(128), 0, stream>>>(
      yh, wph, out, T_SEQ, C_EMB, C_EMB);
}
